// IouHead_4681514353318
// MI455X (gfx1250) — matrix operand order repaired and checked by reading, not compiled, not run on hardware
//
#include <hip/hip_runtime.h>
#include <hip/hip_bf16.h>

typedef __attribute__((ext_vector_type(16))) _Float16 v16h;
typedef __attribute__((ext_vector_type(8)))  float    v8f;

#define NPTS   262144
#define KSEL   4096
#define NBATCH 4
#define LD     4098            // 4096 + 2 halo columns (pad=1 conv)
#define CANDC  2048            // boundary-bin candidate cap per batch

// ---------------- d_out partition (f32) ----------------
// iou_bin  [4,4096,5] : [0,      81920)
// iou_res  [4,4096,1] : [81920,  98304)
// box_sel  [4,4096,7] : [98304,  212992)
// cls_sel  [4,4096,3] : [212992, 262144)
#define OUT_RES 81920
#define OUT_BOX 98304
#define OUT_CLS 212992

__device__ __forceinline__ unsigned ordered_u32(float f) {
    unsigned u = __float_as_uint(f);
    return (u & 0x80000000u) ? ~u : (u | 0x80000000u);
}

// Bitonic sort (descending) of n (power of 2) u64 keys in LDS.
__device__ void bitonic_desc(unsigned long long* d, unsigned n, unsigned tid, unsigned nt) {
    for (unsigned k = 2; k <= n; k <<= 1) {
        for (unsigned j = k >> 1; j > 0; j >>= 1) {
            __syncthreads();
            for (unsigned i = tid; i < n; i += nt) {
                unsigned ix = i ^ j;
                if (ix > i) {
                    unsigned long long a = d[i], b = d[ix];
                    bool up = ((i & k) == 0);           // overall descending
                    if ((a < b) == up) { d[i] = b; d[ix] = a; }
                }
            }
        }
    }
    __syncthreads();
}

// ---------------- K0: zero global histogram + counters ----------------
__global__ void init_kernel(unsigned* __restrict__ ghist, unsigned* __restrict__ ctr) {
    int i = blockIdx.x * blockDim.x + threadIdx.x;
    if (i < NBATCH * 4096) ghist[i] = 0;
    if (i < 32) ctr[i] = 0;
}

// ---------------- K1: ordered keys + 12-bit histogram (full-BW sweep) ----------------
__global__ void keys_hist_kernel(const float* __restrict__ cls,
                                 unsigned* __restrict__ gkeys,
                                 unsigned* __restrict__ ghist) {
    unsigned gsz = gridDim.x * blockDim.x;
    for (unsigned idx = blockIdx.x * blockDim.x + threadIdx.x; idx < NBATCH * NPTS; idx += gsz) {
        unsigned b = idx >> 18;
        const float* cp = cls + (size_t)idx * 3;       // [B][N][3] flat
        float m = fmaxf(fmaxf(cp[0], cp[1]), cp[2]);
        unsigned u = ordered_u32(m);
        gkeys[idx] = u;
        atomicAdd(&ghist[(b << 12) | (u >> 20)], 1u);
    }
}

// ---------------- K2: per-batch threshold bin via parallel suffix scan ----------------
// ctr layout: [0,4)=nab  [4,8)=ncand  [8,12)=T  [12,16)=above  [16,20)=R
__global__ __launch_bounds__(1024) void thresh_kernel(const unsigned* __restrict__ ghist,
                                                      unsigned* __restrict__ ctr) {
    __shared__ unsigned hist[4096];
    __shared__ unsigned ssum[1024];
    const unsigned t = threadIdx.x;
    const int b = blockIdx.x;
    for (unsigned i = t; i < 4096; i += 1024) hist[i] = ghist[(b << 12) + i];
    __syncthreads();
    unsigned s = hist[4 * t] + hist[4 * t + 1] + hist[4 * t + 2] + hist[4 * t + 3];
    ssum[t] = s;
    __syncthreads();
    for (unsigned off = 1; off < 1024; off <<= 1) {     // inclusive suffix scan
        unsigned v = (t + off < 1024) ? ssum[t + off] : 0u;
        __syncthreads();
        ssum[t] += v;
        __syncthreads();
    }
    unsigned acc = (t + 1 < 1024) ? ssum[t + 1] : 0u;   // elements in bins >= 4(t+1)
    for (int q = 3; q >= 0; --q) {                      // walk own 4 bins downward
        unsigned bin = 4 * t + (unsigned)q;
        unsigned c = hist[bin];
        if (acc < KSEL && acc + c >= KSEL) {            // unique crossing
            ctr[8 + b]  = bin;
            ctr[12 + b] = acc;
            ctr[16 + b] = KSEL - acc;
        }
        acc += c;
    }
}

// ---------------- K3: compaction (full-BW sweep of stored keys) ----------------
__global__ void compact_kernel(const unsigned* __restrict__ gkeys,
                               unsigned* __restrict__ ctr,
                               unsigned long long* __restrict__ gsk,
                               unsigned long long* __restrict__ gcand) {
    unsigned gsz = gridDim.x * blockDim.x;
    for (unsigned idx = blockIdx.x * blockDim.x + threadIdx.x; idx < NBATCH * NPTS; idx += gsz) {
        unsigned b = idx >> 18, i = idx & (NPTS - 1);
        unsigned u = gkeys[idx];
        unsigned bin = u >> 20;
        unsigned T = ctr[8 + b];
        if (bin >= T) {
            unsigned long long key =
                ((unsigned long long)u << 32) | (unsigned long long)(0xFFFFFFFFu - i);
            if (bin > T) {
                unsigned p = atomicAdd(&ctr[b], 1u);            // < above < 4096
                gsk[(b << 12) + p] = key;
            } else {
                unsigned p = atomicAdd(&ctr[4 + b], 1u);
                if (p < CANDC) gcand[b * CANDC + p] = key;
            }
        }
    }
}

// ---------------- K4: per-batch in-LDS sort of the 4096 winners -> sel ----------------
__global__ __launch_bounds__(1024) void sortsel_kernel(const unsigned long long* __restrict__ gsk,
                                                       const unsigned long long* __restrict__ gcand,
                                                       const unsigned* __restrict__ ctr,
                                                       int* __restrict__ sel) {
    __shared__ unsigned long long skeys[KSEL];
    __shared__ unsigned long long cand[CANDC];
    const unsigned tid = threadIdx.x;
    const int b = blockIdx.x;
    const unsigned above = ctr[12 + b], R = ctr[16 + b];
    unsigned nc = ctr[4 + b]; if (nc > CANDC) nc = CANDC;

    for (unsigned i = tid; i < KSEL; i += 1024)
        skeys[i] = (i < above) ? gsk[(b << 12) + i] : 0ull;
    for (unsigned i = tid; i < CANDC; i += 1024)
        cand[i] = (i < nc) ? gcand[b * CANDC + i] : 0ull;

    bitonic_desc(cand, CANDC, tid, 1024);               // barrier at entry covers loads
    for (unsigned j = tid; j < R; j += 1024)
        skeys[above + j] = (j < nc) ? cand[j] : 0ull;
    __syncthreads();
    bitonic_desc(skeys, KSEL, tid, 1024);

    for (unsigned k = tid; k < KSEL; k += 1024) {
        unsigned idx = 0xFFFFFFFFu - (unsigned)(skeys[k] & 0xFFFFFFFFull);
        if (idx >= NPTS) idx = 0;
        sel[b * KSEL + k] = (int)idx;
    }
}

// ---------------- Gather: box_sel/cls_sel outputs + f16 conv input w/ halos ----------------
__global__ void gather_kernel(const float* __restrict__ box, const float* __restrict__ cls,
                              const int* __restrict__ sel, float* __restrict__ out,
                              _Float16* __restrict__ xs, _Float16* __restrict__ y1) {
    int t = blockIdx.x * blockDim.x + threadIdx.x;
    if (t >= NBATCH * KSEL) return;
    int b = t >> 12, k = t & 4095;
    int idx = sel[t];
    if (idx < 0) idx = 0; if (idx >= NPTS) idx = NPTS - 1;

    const float* bp = box + ((size_t)b * NPTS + idx) * 7;
    const float* cp = cls + ((size_t)b * NPTS + idx) * 3;
    float* obox = out + OUT_BOX + (size_t)t * 7;
    float* ocls = out + OUT_CLS + (size_t)t * 3;
#pragma unroll
    for (int c = 0; c < 7; ++c) {
        float v = bp[c];
        obox[c] = v;
        xs[((size_t)b * 10 + c) * LD + k + 1] = (_Float16)v;
    }
#pragma unroll
    for (int c = 0; c < 3; ++c) {
        float v = cp[c];
        ocls[c] = v;
        xs[((size_t)b * 10 + 7 + c) * LD + k + 1] = (_Float16)v;
    }
    if (k < 10) {                           // xs halos
        xs[((size_t)b * 10 + k) * LD]          = (_Float16)0.f;
        xs[((size_t)b * 10 + k) * LD + LD - 1] = (_Float16)0.f;
    }
    if (k < 32) {                           // y1 halos (read by conv2 im2col)
        y1[((size_t)b * 32 + k) * LD]          = (_Float16)0.f;
        y1[((size_t)b * 32 + k) * LD + LD - 1] = (_Float16)0.f;
    }
}

// ---------------- Fold BN into f16 weight matrices ----------------
__global__ void prep_kernel(const float* c1w, const float* g1, const float* be1, const float* m1, const float* v1,
                            const float* c2w, const float* g2, const float* be2, const float* m2, const float* v2,
                            const float* binw, const float* binb, const float* resw, const float* resb,
                            _Float16* w1f, float* b1f, _Float16* w2f, float* b2f,
                            _Float16* whf, float* bhf) {
    int tid = threadIdx.x;
    for (int i = tid; i < 32 * 32; i += 256) {
        int c = i >> 5, j = i & 31;
        float inv = g1[c] * rsqrtf(v1[c] + 1e-5f);
        w1f[i] = (j < 30) ? (_Float16)(c1w[c * 30 + j] * inv) : (_Float16)0.f;
    }
    for (int c = tid; c < 32; c += 256) {
        float inv = g1[c] * rsqrtf(v1[c] + 1e-5f);
        b1f[c] = be1[c] - m1[c] * inv;
    }
    for (int i = tid; i < 64 * 96; i += 256) {
        int c = i / 96;
        float inv = g2[c] * rsqrtf(v2[c] + 1e-5f);
        w2f[i] = (_Float16)(c2w[i] * inv);
    }
    for (int c = tid; c < 64; c += 256) {
        float inv = g2[c] * rsqrtf(v2[c] + 1e-5f);
        b2f[c] = be2[c] - m2[c] * inv;
    }
    for (int i = tid; i < 16 * 64; i += 256) {
        int r = i >> 6, k = i & 63;
        float w = 0.f;
        if (r < 5) w = binw[r * 64 + k];
        else if (r == 5) w = resw[k];
        whf[i] = (_Float16)w;
    }
    for (int r = tid; r < 16; r += 256) {
        float bb = 0.f;
        if (r < 5) bb = binb[r]; else if (r == 5) bb = resb[0];
        bhf[r] = bb;
    }
}

// ---------------- k=3 conv as WMMA GEMM (templated, transposed im2col in LDS) ----------------
// out = relu(W @ im2col(in) + bias); in/out f16 halo layout [B][C][LD]
template <int Cin, int Cout>
__global__ __launch_bounds__(32) void conv_wmma_kernel(const _Float16* __restrict__ in,
                                                       const _Float16* __restrict__ W,
                                                       const float* __restrict__ bias,
                                                       _Float16* __restrict__ out) {
    constexpr int Kin  = Cin * 3;
    constexpr int Kpad = (Kin + 31) & ~31;
    constexpr int nK   = Kpad >> 5;
    __shared__ unsigned  rawu[Cin * 9];           // raw tile: Cin rows x 18 cols (f16) as u32
    __shared__ _Float16 tile2[16 * Kpad];         // transposed im2col: [n][j], zero-padded

    const int b  = blockIdx.x >> 8;
    const int l0 = (blockIdx.x & 255) << 4;
    const int lane = threadIdx.x;
    const int n = lane & 15, h = lane >> 4;

    // stage raw activation tile (coalesced b32 global loads; l0 even -> aligned)
    for (int idx = lane; idx < Cin * 9; idx += 32) {
        int row = idx / 9, pr = idx - row * 9;
        rawu[idx] = ((const unsigned*)(in + ((size_t)b * Cin + row) * LD + l0))[pr];
    }
    __syncthreads();
    // build transposed im2col tile: tile2[nn][j] = x[j/3][nn + j%3]; pad rows >= Kin with 0
    {
        const _Float16* raw = (const _Float16*)rawu;
        for (int idx = lane; idx < 16 * Kpad; idx += 32) {
            int nn = idx / Kpad, j = idx - nn * Kpad;
            _Float16 val = (_Float16)0.f;
            if (j < Kin) { int cin = j / 3, t = j - cin * 3; val = raw[cin * 18 + nn + t]; }
            tile2[idx] = val;
        }
    }
    __syncthreads();

    const unsigned* Wu = (const unsigned*)W;      // f16 pairs, 4B-aligned
    const unsigned* Bu = (const unsigned*)tile2;

#pragma unroll
    for (int mt = 0; mt < (Cout >> 4); ++mt) {
        union { v8f v; float f[8]; } acc;
#pragma unroll
        for (int r = 0; r < 8; ++r) acc.f[r] = 0.f;
        const int m = (mt << 4) + n;                    // A row for this lane

#pragma unroll
        for (int ks = 0; ks < nK; ++ks) {
            union { v16h v; unsigned u[8]; } a, bf;
#pragma unroll
            for (int v = 0; v < 8; ++v) {
                // A (16-bit 16x32): lane holds M = lane%16; half h holds K h*16..h*16+15 (contiguous, matches B)
                int kb = h * 16 + (v << 1) + (ks << 5);
                a.u[v] = Wu[(m * Kpad + kb) >> 1];
                // B (16-bit 32x16): half h holds K rows h*16..h*16+15, col = n
                bf.u[v] = Bu[((n * Kpad) >> 1) + (ks << 4) + h * 8 + v];  // contiguous in v
            }
            acc.v = __builtin_amdgcn_wmma_f32_16x16x32_f16(
                false, a.v, false, bf.v, (short)0, acc.v, false, false);
        }
        // C/D layout: lane holds (M = h*8+r, N = n)
#pragma unroll
        for (int r = 0; r < 8; ++r) {
            int row = (mt << 4) + h * 8 + r;
            float y = acc.f[r] + bias[row];
            y = y > 0.f ? y : 0.f;
            out[((size_t)b * Cout + row) * LD + (l0 + 1 + n)] = (_Float16)y;
        }
    }
}

// ---------------- Heads: 16x64 (rows 0-4 bin, row 5 res) @ y2, K=64 ----------------
__global__ __launch_bounds__(32) void heads_kernel(const _Float16* __restrict__ y2,
                                                   const _Float16* __restrict__ W,
                                                   const float* __restrict__ bias,
                                                   float* __restrict__ out) {
    __shared__ unsigned  rawu[64 * 9];            // 64 rows x 18 cols (f16), from even col l0
    __shared__ _Float16 tile2[16 * 64];           // transposed: [n][j]
    const int b  = blockIdx.x >> 8;
    const int l0 = (blockIdx.x & 255) << 4;
    const int lane = threadIdx.x;
    const int n = lane & 15, h = lane >> 4;

    for (int idx = lane; idx < 64 * 9; idx += 32) {
        int row = idx / 9, pr = idx - row * 9;
        rawu[idx] = ((const unsigned*)(y2 + ((size_t)b * 64 + row) * LD + l0))[pr];
    }
    __syncthreads();
    {
        const _Float16* raw = (const _Float16*)rawu;
        for (int idx = lane; idx < 16 * 64; idx += 32) {
            int nn = idx >> 6, j = idx & 63;
            tile2[idx] = raw[j * 18 + nn + 1];    // col l0+1+nn of row j
        }
    }
    __syncthreads();

    const unsigned* Wu = (const unsigned*)W;
    const unsigned* Bu = (const unsigned*)tile2;

    union { v8f v; float f[8]; } acc;
#pragma unroll
    for (int r = 0; r < 8; ++r) acc.f[r] = 0.f;

#pragma unroll
    for (int ks = 0; ks < 2; ++ks) {
        union { v16h v; unsigned u[8]; } a, bf;
#pragma unroll
        for (int v = 0; v < 8; ++v) {
            int kb = h * 16 + (v << 1) + (ks << 5);
            a.u[v]  = Wu[(n * 64 + kb) >> 1];
            bf.u[v] = Bu[n * 32 + (ks << 4) + h * 8 + v];           // contiguous in v
        }
        acc.v = __builtin_amdgcn_wmma_f32_16x16x32_f16(
            false, a.v, false, bf.v, (short)0, acc.v, false, false);
    }
    if (h == 0) {                                       // rows 0..7 live in lanes 0-15
        size_t base = (size_t)b * KSEL + (l0 + n);
#pragma unroll
        for (int r = 0; r < 5; ++r) out[base * 5 + r] = acc.f[r] + bias[r];
        out[OUT_RES + base] = acc.f[5] + bias[5];
    }
}

// ---------------- workspace layout (bytes), total ~8.0 MB ----------------
#define WS_SEL   0u
#define WS_XS    65536u      // f16 [4][10][4098]
#define WS_Y1    393472u     // f16 [4][32][4098]
#define WS_Y2    1442560u    // f16 [4][64][4098]
#define WS_W1F   3540736u
#define WS_B1F   3542784u
#define WS_W2F   3542912u
#define WS_B2F   3555200u
#define WS_WHF   3555456u
#define WS_BHF   3557504u
#define WS_GHIST 3557632u    // u32 [4][4096]
#define WS_CTR   3623168u    // u32 [32]: nab[4] ncand[4] T[4] above[4] R[4]
#define WS_GKEYS 3623808u    // u32 [4][262144]
#define WS_SK64  7818112u    // u64 [4][4096]
#define WS_CAND  7949184u    // u64 [4][2048]  -> end 8014720

extern "C" void kernel_launch(void* const* d_in, const int* in_sizes, int n_in,
                              void* d_out, int out_size, void* d_ws, size_t ws_size,
                              hipStream_t stream) {
    const float* box  = (const float*)d_in[0];
    const float* cls  = (const float*)d_in[1];
    const float* c1w  = (const float*)d_in[2];
    const float* g1   = (const float*)d_in[3];
    const float* be1  = (const float*)d_in[4];
    const float* m1   = (const float*)d_in[5];
    const float* v1   = (const float*)d_in[6];
    const float* c2w  = (const float*)d_in[7];
    const float* g2   = (const float*)d_in[8];
    const float* be2  = (const float*)d_in[9];
    const float* m2   = (const float*)d_in[10];
    const float* v2   = (const float*)d_in[11];
    const float* binw = (const float*)d_in[12];
    const float* binb = (const float*)d_in[13];
    const float* resw = (const float*)d_in[14];
    const float* resb = (const float*)d_in[15];
    float* out = (float*)d_out;
    char* ws = (char*)d_ws;

    int*                sel   = (int*)(ws + WS_SEL);
    _Float16*           xs    = (_Float16*)(ws + WS_XS);
    _Float16*           y1    = (_Float16*)(ws + WS_Y1);
    _Float16*           y2    = (_Float16*)(ws + WS_Y2);
    _Float16*           w1f   = (_Float16*)(ws + WS_W1F);
    float*              b1f   = (float*)(ws + WS_B1F);
    _Float16*           w2f   = (_Float16*)(ws + WS_W2F);
    float*              b2f   = (float*)(ws + WS_B2F);
    _Float16*           whf   = (_Float16*)(ws + WS_WHF);
    float*              bhf   = (float*)(ws + WS_BHF);
    unsigned*           ghist = (unsigned*)(ws + WS_GHIST);
    unsigned*           ctr   = (unsigned*)(ws + WS_CTR);
    unsigned*           gkeys = (unsigned*)(ws + WS_GKEYS);
    unsigned long long* gsk   = (unsigned long long*)(ws + WS_SK64);
    unsigned long long* gcand = (unsigned long long*)(ws + WS_CAND);

    init_kernel<<<64, 256, 0, stream>>>(ghist, ctr);
    keys_hist_kernel<<<256, 256, 0, stream>>>(cls, gkeys, ghist);
    thresh_kernel<<<NBATCH, 1024, 0, stream>>>(ghist, ctr);
    compact_kernel<<<256, 256, 0, stream>>>(gkeys, ctr, gsk, gcand);
    sortsel_kernel<<<NBATCH, 1024, 0, stream>>>(gsk, gcand, ctr, sel);

    prep_kernel<<<1, 256, 0, stream>>>(c1w, g1, be1, m1, v1,
                                       c2w, g2, be2, m2, v2,
                                       binw, binb, resw, resb,
                                       w1f, b1f, w2f, b2f, whf, bhf);
    gather_kernel<<<(NBATCH * KSEL + 255) / 256, 256, 0, stream>>>(box, cls, sel, out, xs, y1);

    conv_wmma_kernel<10, 32><<<NBATCH * 256, 32, 0, stream>>>(xs, w1f, b1f, y1);
    conv_wmma_kernel<32, 64><<<NBATCH * 256, 32, 0, stream>>>(y1, w2f, b2f, y2);
    heads_kernel<<<NBATCH * 256, 32, 0, stream>>>(y2, whf, bhf, out);
}
